// LSTMDecoder_55542517072285
// MI455X (gfx1250) — matrix operand order repaired and checked by reading, not compiled, not run on hardware
//
#include <hip/hip_runtime.h>

// LSTM decoder, B=512, T=64, I=512, H=1024 (gates=4096), O=512.
// Strategy: one-time fp32->bf16 weight conversion into workspace, then per
// timestep: [gates GEMM via v_wmma_f32_16x16x32_bf16] -> [elementwise cell]
// -> [FC GEMM via WMMA, writes fp32 output slice + bf16 x for next step].
// All accumulation in fp32; c-state kept fp32.

#define BB 512   // batch
#define TT 64    // timesteps
#define II 512   // input dim
#define HH 1024  // hidden dim
#define GG 4096  // 4*H gate dim
#define OO 512   // output dim

typedef __bf16 bf16;
typedef __attribute__((ext_vector_type(16))) __bf16 v16bf;
typedef __attribute__((ext_vector_type(8)))  __bf16 v8bf;
typedef __attribute__((ext_vector_type(8)))  float  v8f;

// ---------------- WMMA fragment loaders (CDNA5 wave32 layouts) ----------------

// A-matrix 16x32 bf16, row-major source with leading dim `lda` (elements).
// Lane: M = lane&15. Lanes 0-15 hold K {0..7, 16..23}; lanes 16-31 K {8..15, 24..31}.
__device__ __forceinline__ v16bf load_a_frag(const bf16* __restrict__ A, int lda,
                                             int row, int kBase, int hk) {
  union { v16bf v; v8bf h[2]; } u;
  const bf16* p = A + (size_t)row * lda + kBase + hk * 8;
  u.h[0] = *reinterpret_cast<const v8bf*>(p);        // K = kBase + hk*8 + 0..7
  u.h[1] = *reinterpret_cast<const v8bf*>(p + 16);   // K = kBase + 16 + hk*8 + 0..7
  return u.v;
}

// B-matrix 32x16 bf16 from row-major (N x K) weights: lane = column N,
// lanes 0-15 hold K {0..7, 16..23}; lanes 16-31 K {8..15, 24..31}.
__device__ __forceinline__ v16bf load_b_frag(const bf16* __restrict__ W, int ldw,
                                             int col, int kBase, int hk) {
  union { v16bf v; v8bf h[2]; } u;
  const bf16* p = W + (size_t)col * ldw + kBase + hk * 8;
  u.h[0] = *reinterpret_cast<const v8bf*>(p);
  u.h[1] = *reinterpret_cast<const v8bf*>(p + 16);
  return u.v;
}

// Accumulate a full K range into the 2x4 fragment accumulators of one wave.
__device__ __forceinline__ void mma_block(const bf16* __restrict__ A, int lda,
                                          const bf16* __restrict__ W, int ldw,
                                          int kTotal, int mBase, int nBase,
                                          int hk, int lm, v8f (&acc)[2][4]) {
  for (int kBase = 0; kBase < kTotal; kBase += 32) {
    v16bf a[2], bfr[4];
#pragma unroll
    for (int mi = 0; mi < 2; ++mi)
      a[mi] = load_a_frag(A, lda, mBase + mi * 16 + lm, kBase, hk);
#pragma unroll
    for (int ni = 0; ni < 4; ++ni)
      bfr[ni] = load_b_frag(W, ldw, nBase + ni * 16 + lm, kBase, hk);
#pragma unroll
    for (int mi = 0; mi < 2; ++mi)
#pragma unroll
      for (int ni = 0; ni < 4; ++ni)
        acc[mi][ni] = __builtin_amdgcn_wmma_f32_16x16x32_bf16(
            /*neg_a=*/false, a[mi], /*neg_b=*/false, bfr[ni],
            /*c_mod=*/(short)0, acc[mi][ni], /*reuse_a=*/false, /*reuse_b=*/false);
  }
}

// ---------------- GEMM kernels ----------------

// gates = Xb(BBxII)*Wih^T + Hb(BBxHH)*Whh^T + bias ; gates fp32 (BB x GG)
// grid (GG/256, BB/64), block 256 (8 waves, each 32x64 output)
__global__ __launch_bounds__(256) void gates_gemm_kernel(
    const bf16* __restrict__ Xb, const bf16* __restrict__ Hb,
    const bf16* __restrict__ Wih, const bf16* __restrict__ Whh,
    const float* __restrict__ bias, float* __restrict__ gates) {
  const int lane = threadIdx.x & 31;
  const int hk = lane >> 4;
  const int lm = lane & 15;
  const int wave = threadIdx.x >> 5;
  const int mBase = blockIdx.y * 64 + (wave & 1) * 32;
  const int nBase = blockIdx.x * 256 + (wave >> 1) * 64;

  v8f acc[2][4] = {};
  mma_block(Xb, II, Wih, II, II, mBase, nBase, hk, lm, acc);  // x @ w_ih^T
  mma_block(Hb, HH, Whh, HH, HH, mBase, nBase, hk, lm, acc);  // h @ w_hh^T

#pragma unroll
  for (int mi = 0; mi < 2; ++mi)
#pragma unroll
    for (int ni = 0; ni < 4; ++ni) {
      const int col = nBase + ni * 16 + lm;
      const float bv = bias[col];
      const int row0 = mBase + mi * 16 + hk * 8;  // C layout: VGPR r -> M = hk*8 + r
#pragma unroll
      for (int r = 0; r < 8; ++r)
        gates[(size_t)(row0 + r) * GG + col] = acc[mi][ni][r] + bv;
    }
}

// out_t = Hb(BBxHH)*Wfc^T + b_fc ; writes fp32 slice of (B,T,O) and bf16 x_next
// grid (OO/256, BB/64), block 256
__global__ __launch_bounds__(256) void fc_gemm_kernel(
    const bf16* __restrict__ Hb, const bf16* __restrict__ Wfc,
    const float* __restrict__ bfc, float* __restrict__ out,
    bf16* __restrict__ Xb, int t) {
  const int lane = threadIdx.x & 31;
  const int hk = lane >> 4;
  const int lm = lane & 15;
  const int wave = threadIdx.x >> 5;
  const int mBase = blockIdx.y * 64 + (wave & 1) * 32;
  const int nBase = blockIdx.x * 256 + (wave >> 1) * 64;

  v8f acc[2][4] = {};
  mma_block(Hb, HH, Wfc, HH, HH, mBase, nBase, hk, lm, acc);

#pragma unroll
  for (int mi = 0; mi < 2; ++mi)
#pragma unroll
    for (int ni = 0; ni < 4; ++ni) {
      const int col = nBase + ni * 16 + lm;
      const float bv = bfc[col];
      const int row0 = mBase + mi * 16 + hk * 8;
#pragma unroll
      for (int r = 0; r < 8; ++r) {
        const float v = acc[mi][ni][r] + bv;
        out[(size_t)(row0 + r) * (TT * OO) + (size_t)t * OO + col] = v;  // (B,T,O)
        Xb[(size_t)(row0 + r) * OO + col] = (bf16)v;                     // next-step x
      }
    }
}

// ---------------- elementwise LSTM cell ----------------

__device__ __forceinline__ float sigmoid_f(float x) {
  return 1.0f / (1.0f + __expf(-x));
}
__device__ __forceinline__ float tanh_f(float x) {
  const float e = __expf(-2.0f * x);
  return (1.0f - e) / (1.0f + e);
}

// grid BB*HH/256, block 256
__global__ __launch_bounds__(256) void lstm_cell_kernel(
    const float* __restrict__ gates, float* __restrict__ c,
    bf16* __restrict__ Hb) {
  const int idx = blockIdx.x * 256 + threadIdx.x;  // [0, BB*HH)
  const int b = idx >> 10;                          // HH == 1024
  const int n = idx & (HH - 1);
  const float* g = gates + (size_t)b * GG;
  const float iv = sigmoid_f(g[n]);
  const float fv = sigmoid_f(g[HH + n]);
  const float gv = tanh_f(g[2 * HH + n]);
  const float ov = sigmoid_f(g[3 * HH + n]);
  const float cv = fv * c[idx] + iv * gv;
  c[idx] = cv;
  Hb[idx] = (bf16)(ov * tanh_f(cv));
}

// ---------------- prep kernels ----------------

__global__ void f2bf_kernel(bf16* __restrict__ dst, const float* __restrict__ src, int n) {
  const int i = blockIdx.x * blockDim.x + threadIdx.x;
  if (i < n) dst[i] = (bf16)src[i];
}
__global__ void bias_sum_kernel(float* __restrict__ dst, const float* __restrict__ a,
                                const float* __restrict__ b, int n) {
  const int i = blockIdx.x * blockDim.x + threadIdx.x;
  if (i < n) dst[i] = a[i] + b[i];
}
__global__ void copyf_kernel(float* __restrict__ dst, const float* __restrict__ src, int n) {
  const int i = blockIdx.x * blockDim.x + threadIdx.x;
  if (i < n) dst[i] = src[i];
}

// ---------------- host launch ----------------

extern "C" void kernel_launch(void* const* d_in, const int* in_sizes, int n_in,
                              void* d_out, int out_size, void* d_ws, size_t ws_size,
                              hipStream_t stream) {
  (void)in_sizes; (void)n_in; (void)out_size; (void)ws_size;
  const float* inputs = (const float*)d_in[0];  // (B,1,I)
  const float* hidden = (const float*)d_in[1];  // (1,B,H)
  const float* cell   = (const float*)d_in[2];  // (1,B,H)
  const float* w_ih   = (const float*)d_in[3];  // (4H,I)
  const float* w_hh   = (const float*)d_in[4];  // (4H,H)
  const float* b_ih   = (const float*)d_in[5];  // (4H)
  const float* b_hh   = (const float*)d_in[6];  // (4H)
  const float* w_fc   = (const float*)d_in[7];  // (O,H)
  const float* b_fc   = (const float*)d_in[8];  // (O)
  float* out = (float*)d_out;                   // (B,T,O)

  // Workspace carve (256B aligned): ~26 MB total.
  size_t off = 0;
  char* ws = (char*)d_ws;
  auto carve = [&](size_t bytes) -> void* {
    void* p = ws + off;
    off += (bytes + 255) & ~(size_t)255;
    return p;
  };
  bf16*  wih_b = (bf16*) carve((size_t)GG * II * sizeof(bf16));
  bf16*  whh_b = (bf16*) carve((size_t)GG * HH * sizeof(bf16));
  bf16*  wfc_b = (bf16*) carve((size_t)OO * HH * sizeof(bf16));
  float* biasv = (float*)carve((size_t)GG * sizeof(float));
  float* gates = (float*)carve((size_t)BB * GG * sizeof(float));
  bf16*  xb    = (bf16*) carve((size_t)BB * II * sizeof(bf16));
  bf16*  hb    = (bf16*) carve((size_t)BB * HH * sizeof(bf16));
  float* cst   = (float*)carve((size_t)BB * HH * sizeof(float));

  const int thr = 256;
  // One-time conversions (re-done every call: deterministic, inputs never mutated).
  f2bf_kernel<<<(GG * II + thr - 1) / thr, thr, 0, stream>>>(wih_b, w_ih, GG * II);
  f2bf_kernel<<<(GG * HH + thr - 1) / thr, thr, 0, stream>>>(whh_b, w_hh, GG * HH);
  f2bf_kernel<<<(OO * HH + thr - 1) / thr, thr, 0, stream>>>(wfc_b, w_fc, OO * HH);
  bias_sum_kernel<<<(GG + thr - 1) / thr, thr, 0, stream>>>(biasv, b_ih, b_hh, GG);
  f2bf_kernel<<<(BB * II + thr - 1) / thr, thr, 0, stream>>>(xb, inputs, BB * II);
  f2bf_kernel<<<(BB * HH + thr - 1) / thr, thr, 0, stream>>>(hb, hidden, BB * HH);
  copyf_kernel<<<(BB * HH + thr - 1) / thr, thr, 0, stream>>>(cst, cell, BB * HH);

  const dim3 gGrid(GG / 256, BB / 64);  // (16, 8)
  const dim3 fGrid(OO / 256, BB / 64);  // (2, 8)
  for (int t = 0; t < TT; ++t) {
    gates_gemm_kernel<<<gGrid, thr, 0, stream>>>(xb, hb, wih_b, whh_b, biasv, gates);
    lstm_cell_kernel<<<(BB * HH) / thr, thr, 0, stream>>>(gates, cst, hb);
    fc_gemm_kernel<<<fGrid, thr, 0, stream>>>(hb, wfc_b, b_fc, out, xb, t);
  }
}
